// MultiHeadAttention_13855564497021
// MI455X (gfx1250) — matrix operand order repaired and checked by reading, not compiled, not run on hardware
//
#include <hip/hip_runtime.h>

// ---------------------------------------------------------------------------
// MHA forward for MI455X (gfx1250, wave32).
//   BATCH=8, LEN=1024, DIM=1024, HEAD=16, DH=64
// bf16 v_wmma_f32_16x16x32_bf16 everywhere, f32 accumulate + f32 softmax.
// All f32->bf16 conversion is done ONCE by a bandwidth-bound pre-pass using
// v_cvt_pk_bf16_f32; GEMM inner loops are pure b128 loads + wmma.
// ---------------------------------------------------------------------------

#define BATCH 8
#define SEQ   1024
#define DIMSZ 1024
#define NH    16
#define DH    64

typedef __bf16 bf16_t;
typedef bf16_t v16bf __attribute__((ext_vector_type(16)));
typedef float  v8f   __attribute__((ext_vector_type(8)));

struct Frag { union { unsigned u[8]; uint4 q[2]; v16bf v; }; };

// packed f32x2 -> bf16x2 (RNE) via the hardware converter
__device__ __forceinline__ unsigned cvtpk(float lo, float hi) {
    unsigned r;
    asm("v_cvt_pk_bf16_f32 %0, %1, %2" : "=v"(r) : "v"(lo), "v"(hi));
    return r;
}
// scalar f32 -> bf16 bits (used only for 8 stores per tile; cold path)
__device__ __forceinline__ unsigned short bf16r(float f) {
    unsigned u = __float_as_uint(f);
    return (unsigned short)((u + 0x7FFFu + ((u >> 16) & 1u)) >> 16);
}

__device__ __forceinline__ v8f wmma_bf16(const Frag& a, const Frag& b, v8f c) {
    return __builtin_amdgcn_wmma_f32_16x16x32_bf16(
        false, a.v, false, b.v, (short)0, c, false, false);
}

// ---------------------------------------------------------------------------
// Kernel 0: f32 -> bf16 bulk convert (8 elements / thread, b128 in, b128 out)
// ---------------------------------------------------------------------------
__global__ __launch_bounds__(256)
void cvt_kernel(const float* __restrict__ in, unsigned short* __restrict__ out,
                int n8) {
    const int idx = blockIdx.x * 256 + threadIdx.x;
    if (idx >= n8) return;
    const float4* p = (const float4*)in + (size_t)idx * 2;
    const float4 f0 = p[0], f1 = p[1];
    uint4 o;
    o.x = cvtpk(f0.x, f0.y);
    o.y = cvtpk(f0.z, f0.w);
    o.z = cvtpk(f1.x, f1.y);
    o.w = cvtpk(f1.z, f1.w);
    ((uint4*)out)[idx] = o;
}

// ---------------------------------------------------------------------------
// Kernel 1: projection GEMM  y[m,n] = sum_k x[m,k] * W[n,k]  (bf16 in, f32 acc)
// Each wave: 2x2 tiles of 16x16 (4 wmma / K-step, 8 b128 loads / K-step).
// Block = 4 waves arranged 2x2 -> 64x64 per block. grid = (M/64, N/64).
//   mode 0: y -> out[((b*NH+h)*SEQ + s)*DH + d]      (Q, K)
//   mode 1: y -> out[((b*NH+h)*DH  + d)*SEQ + s]     (V, transposed)
// ---------------------------------------------------------------------------
__global__ __launch_bounds__(128)
void proj_kernel(const unsigned short* __restrict__ Xb,
                 const unsigned short* __restrict__ Wb,
                 unsigned short* __restrict__ out, int mode) {
    const int lane = threadIdx.x & 31;
    const int wv   = threadIdx.x >> 5;
    const int mt0  = blockIdx.x * 4 + (wv & 1) * 2;   // first of 2 m-tiles
    const int nt0  = blockIdx.y * 4 + (wv >> 1) * 2;  // first of 2 n-tiles

    const int l15  = lane & 15;
    const int aOff = (lane & 16) ? 4 : 0;             // dword offset, A layout
    const int bOff = (lane & 16) ? 4 : 0;             // dword offset, B layout

    // per-lane row pointers (dword granularity; DIM bf16 = 512 dwords/row)
    const unsigned* __restrict__ A0 = (const unsigned*)Xb + (size_t)(mt0 * 16 + l15) * 512;
    const unsigned* __restrict__ A1 = A0 + 16 * 512;
    const unsigned* __restrict__ B0 = (const unsigned*)Wb + (size_t)(nt0 * 16 + l15) * 512;
    const unsigned* __restrict__ B1 = B0 + 16 * 512;

    v8f c00 = {0.f,0.f,0.f,0.f,0.f,0.f,0.f,0.f};
    v8f c01 = c00, c10 = c00, c11 = c00;

    for (int kk = 0; kk < DIMSZ / 32; ++kk) {
        const int kd = kk * 16;                       // dword base of K-step
        Frag a0, a1, b0, b1;
        a0.q[0] = *(const uint4*)(A0 + kd + aOff);
        a0.q[1] = *(const uint4*)(A0 + kd + aOff + 8);
        a1.q[0] = *(const uint4*)(A1 + kd + aOff);
        a1.q[1] = *(const uint4*)(A1 + kd + aOff + 8);
        b0.q[0] = *(const uint4*)(B0 + kd + bOff);
        b0.q[1] = *(const uint4*)(B0 + kd + bOff + 8);
        b1.q[0] = *(const uint4*)(B1 + kd + bOff);
        b1.q[1] = *(const uint4*)(B1 + kd + bOff + 8);
        c00 = wmma_bf16(a0, b0, c00);
        c01 = wmma_bf16(a0, b1, c01);
        c10 = wmma_bf16(a1, b0, c10);
        c11 = wmma_bf16(a1, b1, c11);
    }

    // store: lane col = ntile*16 + l15 ; VGPR i -> row i + rHi of the tile
    const int rHi = (lane & 16) ? 8 : 0;
    const v8f* accs[4] = {&c00, &c01, &c10, &c11};
#pragma unroll
    for (int t = 0; t < 4; ++t) {
        const int mt = mt0 + (t >> 1);
        const int nc = (nt0 + (t & 1)) * 16 + l15;
        const int h = nc >> 6, d = nc & 63;
        const v8f& c = *accs[t];
#pragma unroll
        for (int i = 0; i < 8; ++i) {
            const int g = mt * 16 + i + rHi;          // global row in [0, B*SEQ)
            const int bb = g >> 10, s = g & 1023;
            const size_t o = (mode == 0)
                ? (((size_t)bb * NH + h) * SEQ + s) * DH + d
                : (((size_t)bb * NH + h) * DH + d) * SEQ + s;
            out[o] = bf16r(c[i]);
        }
    }
}

// ---------------------------------------------------------------------------
// Kernel 2: fused  S = Q K^T / 8 ; P = softmax(S) ; O = P V
// One workgroup (4 waves) per (b, h, 16-row tile). S tile 16x1024 f32 in LDS.
// grid = B*NH*(SEQ/16) = 8192 , block = 128
// ---------------------------------------------------------------------------
__global__ __launch_bounds__(128)
void attn_kernel(const unsigned short* __restrict__ q_ws,
                 const unsigned short* __restrict__ k_ws,
                 const unsigned short* __restrict__ v_ws,
                 float* __restrict__ out, float* __restrict__ att) {
    __shared__ float P[16 * 1024];                    // 64 KB score/prob tile

    const int tid  = threadIdx.x;
    const int lane = tid & 31;
    const int wv   = tid >> 5;
    const int l15  = lane & 15;

    const int stile = blockIdx.x & 63;
    const size_t bh = blockIdx.x >> 6;
    const int bb = (int)(bh >> 4);
    const int h  = (int)(bh & 15);

    const unsigned* __restrict__ Q = (const unsigned*)(q_ws + bh * (size_t)SEQ * DH);
    const unsigned* __restrict__ K = (const unsigned*)(k_ws + bh * (size_t)SEQ * DH);
    const unsigned* __restrict__ V = (const unsigned*)(v_ws + bh * (size_t)DH * SEQ);

    const int aOff = (lane & 16) ? 4 : 0;             // dword offsets per ISA layouts
    const int bOff = (lane & 16) ? 4 : 0;
    const int rHi  = (lane & 16) ? 8 : 0;

    // ---- Phase 1: scores (d = 64 -> 2 wmma K-steps per column tile) ----
    Frag a0, a1;
    {
        // Q row = DH bf16 = 32 dwords
        const unsigned* __restrict__ Qr = Q + (size_t)(stile * 16 + l15) * 32;
        a0.q[0] = *(const uint4*)(Qr + aOff);
        a0.q[1] = *(const uint4*)(Qr + aOff + 8);
        a1.q[0] = *(const uint4*)(Qr + aOff + 16);
        a1.q[1] = *(const uint4*)(Qr + aOff + 24);
    }
    for (int j = 0; j < 16; ++j) {                    // wave handles 16 col tiles
        const int tcol = (wv * 16 + j) * 16 + l15;
        const unsigned* __restrict__ Kr = K + (size_t)tcol * 32;
        Frag b0, b1;
        b0.q[0] = *(const uint4*)(Kr + bOff);
        b0.q[1] = *(const uint4*)(Kr + bOff + 8);
        b1.q[0] = *(const uint4*)(Kr + bOff + 16);
        b1.q[1] = *(const uint4*)(Kr + bOff + 24);
        v8f c = {0.f,0.f,0.f,0.f,0.f,0.f,0.f,0.f};
        c = wmma_bf16(a0, b0, c);
        c = wmma_bf16(a1, b1, c);
#pragma unroll
        for (int i = 0; i < 8; ++i)
            P[(i + rHi) * 1024 + tcol] = c[i] * 0.125f;   // 1/sqrt(64)
    }
    __syncthreads();

    // ---- Phase 2: f32 softmax, 8 lanes per row, shuffle reductions ----
    {
        const int row = tid >> 3;
        const int sub = tid & 7;
        float* __restrict__ Pr = P + row * 1024;
        const int off = sub * 128;

        float mx = -1e30f;
        for (int j2 = 0; j2 < 128; ++j2) mx = fmaxf(mx, Pr[off + j2]);
        for (int o = 4; o > 0; o >>= 1) mx = fmaxf(mx, __shfl_xor(mx, o, 8));

        float sum = 0.f;
        for (int j2 = 0; j2 < 128; ++j2) {
            const float e = __expf(Pr[off + j2] - mx);
            Pr[off + j2] = e;
            sum += e;
        }
        for (int o = 4; o > 0; o >>= 1) sum += __shfl_xor(sum, o, 8);
        const float inv = 1.f / sum;

        float* __restrict__ ag =
            att + ((bh * SEQ) + (size_t)(stile * 16 + row)) * SEQ + off;
        for (int j2 = 0; j2 < 128; ++j2) {
            const float w = Pr[off + j2] * inv;
            Pr[off + j2] = w;                         // keep for P@V
            ag[j2] = w;                               // emit att output
        }
    }
    __syncthreads();

    // ---- Phase 3: O[16x64] = P[16x1024] @ V[1024x64]; wave wv -> 16 cols ----
    {
        const int dcol = wv * 16 + l15;               // d within head
        const unsigned* __restrict__ Vc = V + (size_t)dcol * 512;  // SEQ bf16 = 512 dw
        const float* __restrict__ Pr = P + l15 * 1024;
        v8f c = {0.f,0.f,0.f,0.f,0.f,0.f,0.f,0.f};
        for (int kk = 0; kk < SEQ / 32; ++kk) {
            Frag a, b;
            // A from LDS f32: two 8-float chunks (k-halves), pack with cvt_pk
            const float4 fa = *(const float4*)(Pr + kk * 32 + 2 * aOff);
            const float4 fb = *(const float4*)(Pr + kk * 32 + 2 * aOff + 4);
            const float4 fc = *(const float4*)(Pr + kk * 32 + 16 + 2 * aOff);
            const float4 fd = *(const float4*)(Pr + kk * 32 + 16 + 2 * aOff + 4);
            a.u[0] = cvtpk(fa.x, fa.y);  a.u[1] = cvtpk(fa.z, fa.w);
            a.u[2] = cvtpk(fb.x, fb.y);  a.u[3] = cvtpk(fb.z, fb.w);
            a.u[4] = cvtpk(fc.x, fc.y);  a.u[5] = cvtpk(fc.z, fc.w);
            a.u[6] = cvtpk(fd.x, fd.y);  a.u[7] = cvtpk(fd.z, fd.w);
            b.q[0] = *(const uint4*)(Vc + kk * 16 + bOff);
            b.q[1] = *(const uint4*)(Vc + kk * 16 + bOff + 8);
            c = wmma_bf16(a, b, c);
        }
#pragma unroll
        for (int i = 0; i < 8; ++i) {
            const int s = stile * 16 + i + rHi;
            out[((size_t)bb * SEQ + s) * DIMSZ + h * DH + dcol] = c[i];
        }
    }
}

// ---------------------------------------------------------------------------
extern "C" void kernel_launch(void* const* d_in, const int* in_sizes, int n_in,
                              void* d_out, int out_size, void* d_ws, size_t ws_size,
                              hipStream_t stream) {
    const float* qry = (const float*)d_in[0];
    const float* key = (const float*)d_in[1];
    const float* val = (const float*)d_in[2];
    const float* w_q = (const float*)d_in[3];
    const float* w_k = (const float*)d_in[4];
    const float* w_v = (const float*)d_in[5];

    const size_t xN = (size_t)BATCH * SEQ * DIMSZ;    // 8,388,608
    const size_t wN = (size_t)DIMSZ * DIMSZ;          // 1,048,576

    unsigned short* p = (unsigned short*)d_ws;
    unsigned short* xb_q = p;  p += xN;               // bf16 activations
    unsigned short* xb_k = p;  p += xN;
    unsigned short* xb_v = p;  p += xN;
    unsigned short* wb_q = p;  p += wN;               // bf16 weights
    unsigned short* wb_k = p;  p += wN;
    unsigned short* wb_v = p;  p += wN;
    unsigned short* q_ws = p;  p += xN;               // projected (bf16)
    unsigned short* k_ws = p;  p += xN;
    unsigned short* v_ws = p;  p += xN;               // transposed [b,h,d,s]

    float* outp = (float*)d_out;                      // [B, SEQ, DIM]
    float* attp = outp + xN;                          // [B, NH, SEQ, SEQ]

    const int xBlocks = (int)(xN / 8 / 256);          // 4096
    const int wBlocks = (int)(wN / 8 / 256);          // 512
    cvt_kernel<<<xBlocks, 256, 0, stream>>>(qry, xb_q, (int)(xN / 8));
    cvt_kernel<<<xBlocks, 256, 0, stream>>>(key, xb_k, (int)(xN / 8));
    cvt_kernel<<<xBlocks, 256, 0, stream>>>(val, xb_v, (int)(xN / 8));
    cvt_kernel<<<wBlocks, 256, 0, stream>>>(w_q, wb_q, (int)(wN / 8));
    cvt_kernel<<<wBlocks, 256, 0, stream>>>(w_k, wb_k, (int)(wN / 8));
    cvt_kernel<<<wBlocks, 256, 0, stream>>>(w_v, wb_v, (int)(wN / 8));

    dim3 gproj(BATCH * SEQ / 64, DIMSZ / 64);         // (128, 16)
    proj_kernel<<<gproj, 128, 0, stream>>>(xb_q, wb_q, q_ws, 0);
    proj_kernel<<<gproj, 128, 0, stream>>>(xb_k, wb_k, k_ws, 0);
    proj_kernel<<<gproj, 128, 0, stream>>>(xb_v, wb_v, v_ws, 1);

    attn_kernel<<<BATCH * NH * (SEQ / 16), 128, 0, stream>>>(
        q_ws, k_ws, v_ws, outp, attp);
}
